// JAX_SCF_37675453120639
// MI455X (gfx1250) — matrix operand order repaired and checked by reading, compiled, not run on hardware
//
#include <hip/hip_runtime.h>

typedef __attribute__((ext_vector_type(16))) __bf16 v16bf;
typedef __attribute__((ext_vector_type(8)))  float  v8f;
typedef __attribute__((ext_vector_type(4)))  float  f4;

#define N_COORDS 131072
#define NAO_CART 280
#define NAO_SPH  260
#define KPAD     288   // 9 * 32
#define NPAD     272   // 17 * 16
#define MTILE    64
#define NTILES   17
#define KSTEPS   9

struct bf16x16 { f4 lo, hi; };

// ---------------------------------------------------------------------------
// Prep: Bt[n][k] = cart2sph[k][n] as bf16, zero padded to [272][288].
// cart2sph is [280][260] row-major f32.
// ---------------------------------------------------------------------------
__global__ void prep_b(const float* __restrict__ c2s, __bf16* __restrict__ Bt) {
    int i = blockIdx.x * 256 + threadIdx.x;
    if (i >= NPAD * KPAD) return;
    int n = i / KPAD;
    int k = i - n * KPAD;
    float v = (n < NAO_SPH && k < NAO_CART) ? c2s[k * NAO_SPH + n] : 0.0f;
    Bt[i] = (__bf16)v;
}

// ---------------------------------------------------------------------------
// Fused AO-eval + GEMM.
// Block: 128 threads (4 wave32). M-tile: 64 rows.
// Shell bookkeeping (compile-time constants from the reference):
//   l=0 group: j=0..39, shell id = (j>>1)*5 + (j&1), atom = j>>1, K = j
//   l=1 group: j=0..39, shell id = (j>>1)*5 + 2 + (j&1), atom = j>>1, K = 40+3j
//   l=2 group: j=0..19, shell id = 5j + 4, atom = j, K = 160+6j
//   anorms: l<=1 -> 1; l=2 -> {1/sqrt3, 1, 1, 1/sqrt3, 1, 1}
// ---------------------------------------------------------------------------
__global__ __launch_bounds__(128)
void eval_ao_gemm(const float* __restrict__ coords,
                  const float* __restrict__ centers,
                  const float* __restrict__ zetas,
                  const float* __restrict__ coeffs,
                  const float* __restrict__ norm,
                  const __bf16* __restrict__ Bt,
                  float* __restrict__ out) {
    __shared__ __bf16 A[MTILE * KPAD];   // 36 KB

    const int tid  = threadIdx.x;
    const int r0   = blockIdx.x * MTILE;
    const int rl   = tid & 63;           // local row
    const int half = tid >> 6;           // shell partition
    const int row  = r0 + rl;

    const float px = coords[row * 3 + 0];
    const float py = coords[row * 3 + 1];
    const float pz = coords[row * 3 + 2];
    const float IS3 = 0.57735026918962576f;  // 1/sqrt(3)

    auto radial = [&](int sid, float rs2) -> float {
        float rad = 0.0f;
#pragma unroll
        for (int p = 0; p < 8; ++p)
            rad += coeffs[sid * 8 + p] * __expf(-zetas[sid * 8 + p] * rs2);
        return rad;
    };

    if (half == 0) {
        // l = 0 shells: K = 0..39
        for (int j = 0; j < 40; ++j) {
            int sid = (j >> 1) * 5 + (j & 1);
            int atom = j >> 1;
            float dx = px - centers[atom * 3 + 0];
            float dy = py - centers[atom * 3 + 1];
            float dz = pz - centers[atom * 3 + 2];
            float rs2 = dx * dx + dy * dy + dz * dz;
            float rad = radial(sid, rs2);
            A[rl * KPAD + j] = (__bf16)(rad * norm[j]);
        }
        // l = 1 shells j = 0..9: K = 40..69
        for (int j = 0; j < 10; ++j) {
            int sid = (j >> 1) * 5 + 2 + (j & 1);
            int atom = j >> 1;
            float dx = px - centers[atom * 3 + 0];
            float dy = py - centers[atom * 3 + 1];
            float dz = pz - centers[atom * 3 + 2];
            float rs2 = dx * dx + dy * dy + dz * dz;
            float rad = radial(sid, rs2);
            int K = 40 + 3 * j;
            A[rl * KPAD + K + 0] = (__bf16)(dx * rad * norm[K + 0]);
            A[rl * KPAD + K + 1] = (__bf16)(dy * rad * norm[K + 1]);
            A[rl * KPAD + K + 2] = (__bf16)(dz * rad * norm[K + 2]);
        }
    } else {
        // l = 1 shells j = 10..39: K = 70..159
        for (int j = 10; j < 40; ++j) {
            int sid = (j >> 1) * 5 + 2 + (j & 1);
            int atom = j >> 1;
            float dx = px - centers[atom * 3 + 0];
            float dy = py - centers[atom * 3 + 1];
            float dz = pz - centers[atom * 3 + 2];
            float rs2 = dx * dx + dy * dy + dz * dz;
            float rad = radial(sid, rs2);
            int K = 40 + 3 * j;
            A[rl * KPAD + K + 0] = (__bf16)(dx * rad * norm[K + 0]);
            A[rl * KPAD + K + 1] = (__bf16)(dy * rad * norm[K + 1]);
            A[rl * KPAD + K + 2] = (__bf16)(dz * rad * norm[K + 2]);
        }
        // l = 2 shells: K = 160..279
        for (int j = 0; j < 20; ++j) {
            int sid = 5 * j + 4;
            int atom = j;
            float dx = px - centers[atom * 3 + 0];
            float dy = py - centers[atom * 3 + 1];
            float dz = pz - centers[atom * 3 + 2];
            float rs2 = dx * dx + dy * dy + dz * dz;
            float rad = radial(sid, rs2);
            int K = 160 + 6 * j;
            A[rl * KPAD + K + 0] = (__bf16)(dx * dx * IS3 * rad * norm[K + 0]);
            A[rl * KPAD + K + 1] = (__bf16)(dx * dy * rad * norm[K + 1]);
            A[rl * KPAD + K + 2] = (__bf16)(dx * dz * rad * norm[K + 2]);
            A[rl * KPAD + K + 3] = (__bf16)(dy * dy * IS3 * rad * norm[K + 3]);
            A[rl * KPAD + K + 4] = (__bf16)(dy * dz * rad * norm[K + 4]);
            A[rl * KPAD + K + 5] = (__bf16)(dz * dz * rad * norm[K + 5]);
        }
        // zero pad K = 280..287
#pragma unroll
        for (int k = NAO_CART; k < KPAD; ++k)
            A[rl * KPAD + k] = (__bf16)0.0f;
    }

    __syncthreads();

    // ---------------- WMMA GEMM stage ----------------
    const int lane = tid & 31;
    const int w    = tid >> 5;           // wave id: rows 16w..16w+15
    const int lm   = lane & 15;
    const int kh   = lane >> 4;          // K-half select
    const int mrow = 16 * w + lm;

    // Preload A fragments for all 9 K-steps.
    // 16-bit A 16x32 layout: lane m=lm; halves e<8 -> k = kb + 8*kh + e,
    //                        halves e>=8 -> k = kb + 16 + 8*kh + (e-8).
    v16bf afr[KSTEPS];
#pragma unroll
    for (int ks = 0; ks < KSTEPS; ++ks) {
        int kb = 32 * ks;
        f4 lo = *(const f4*)&A[mrow * KPAD + kb + 8 * kh];
        f4 hi = *(const f4*)&A[mrow * KPAD + kb + 16 + 8 * kh];
        bf16x16 t{lo, hi};
        afr[ks] = __builtin_bit_cast(v16bf, t);
    }

    // B 32x16 layout: lane n = lm; e<8 -> k = kb + 8*kh + e, e>=8 -> k = kb + 16 + 8*kh + (e-8).
    const int kbB = 8 * kh;
    for (int nt = 0; nt < NTILES; ++nt) {
        const __bf16* bp = Bt + (nt * 16 + lm) * KPAD + kbB;
        if (nt + 1 < NTILES)
            __builtin_prefetch(Bt + ((nt + 1) * 16 + lm) * KPAD + kbB, 0, 1);
        v8f acc = {};
#pragma unroll
        for (int ks = 0; ks < KSTEPS; ++ks) {
            f4 lo = *(const f4*)(bp + 32 * ks);
            f4 hi = *(const f4*)(bp + 32 * ks + 16);
            bf16x16 t{lo, hi};
            v16bf b = __builtin_bit_cast(v16bf, t);
            acc = __builtin_amdgcn_wmma_f32_16x16x32_bf16(
                false, afr[ks], false, b, (short)0, acc, false, false);
        }
        // D layout: VGPR j -> row (j + 8*kh), col = lm within tile.
        int col = nt * 16 + lm;
        if (col < NAO_SPH) {
#pragma unroll
            for (int j = 0; j < 8; ++j)
                out[(r0 + 16 * w + j + 8 * kh) * NAO_SPH + col] = acc[j];
        }
    }
}

extern "C" void kernel_launch(void* const* d_in, const int* in_sizes, int n_in,
                              void* d_out, int out_size, void* d_ws, size_t ws_size,
                              hipStream_t stream) {
    const float* coords  = (const float*)d_in[0];
    const float* centers = (const float*)d_in[1];
    const float* zetas   = (const float*)d_in[2];
    const float* coeffs  = (const float*)d_in[3];
    const float* norm    = (const float*)d_in[4];
    const float* c2s     = (const float*)d_in[5];
    __bf16* Bt = (__bf16*)d_ws;          // 272*288*2 B = 153 KB scratch
    float* out = (float*)d_out;

    int prep_elems = NPAD * KPAD;
    prep_b<<<(prep_elems + 255) / 256, 256, 0, stream>>>(c2s, Bt);
    eval_ao_gemm<<<N_COORDS / MTILE, 128, 0, stream>>>(
        coords, centers, zetas, coeffs, norm, Bt, out);
}
